// Attention_16449724744497
// MI455X (gfx1250) — matrix operand order repaired and checked by reading, not compiled, not run on hardware
//
#include <hip/hip_runtime.h>
#include <hip/hip_bf16.h>
#include <stdint.h>

#define B_   256
#define N_   196
#define H_   12
#define C_   768
#define DH   64
#define NPAD 208    // 13*16
#define KPAD 224    // 7*32
#define BH   (B_*H_)      // 3072
#define ROWS (B_*N_)      // 50176
#define RT   (ROWS/16)    // 3136

typedef __attribute__((ext_vector_type(16))) __bf16 bf16x16;
typedef __attribute__((ext_vector_type(8)))  float  f32x8;
typedef __attribute__((ext_vector_type(4)))  unsigned uint32x4;
typedef __attribute__((ext_vector_type(8)))  int      int32x8;
typedef __attribute__((ext_vector_type(4)))  int      int32x4;

union FragBF { bf16x16 v; uint4 q[2]; };

__device__ __forceinline__ f32x8 wmma_bf16(const FragBF& a, const FragBF& b, f32x8 c) {
  return __builtin_amdgcn_wmma_f32_16x16x32_bf16(false, a.v, false, b.v, (short)0, c, false, false);
}

// TDM: flat 1-D tile of `nelem` 2-byte elements, global -> LDS. D# per CDNA5 ISA ch.8.
__device__ __forceinline__ void tdm_load_1d_bf16(void* lds, const void* gmem, unsigned nelem) {
  unsigned lds_addr = (unsigned)(uintptr_t)lds;            // aperture maps addr[31:0] -> LDS
  unsigned long long ga = (unsigned long long)(uintptr_t)gmem;
  uint32x4 g0;
  g0[0] = 1u;                                              // count=1, user mode
  g0[1] = lds_addr;                                        // lds_addr [63:32]
  g0[2] = (unsigned)ga;                                    // global_addr[31:0]
  g0[3] = (unsigned)((ga >> 32) & 0x01FFFFFFu) | (2u << 30); // global_addr[56:32] | type=2
  int32x8 g1;
  g1[0] = (int)(1u << 16);                                 // data_size=1 (2 bytes)
  g1[1] = (int)((nelem & 0xFFFFu) << 16);                  // tensor_dim0[15:0] @ bits 63:48
  g1[2] = (int)(((nelem >> 16) & 0xFFFFu) | (1u << 16));   // tensor_dim0[31:16], tensor_dim1=1
  g1[3] = (int)((nelem & 0xFFFFu) << 16);                  // tile_dim0 @ bits 127:112 (<=65535)
  g1[4] = 0;                                               // tile_dim1=0, tile_dim2=0
  g1[5] = (int)nelem;                                      // tensor_dim0_stride[31:0]
  g1[6] = 0;                                               // stride hi, dim1_stride lo
  g1[7] = 0;
  int32x4 z4 = {0, 0, 0, 0};
  int32x8 z8 = {0, 0, 0, 0, 0, 0, 0, 0};
  __builtin_amdgcn_tensor_load_to_lds(g0, g1, z4, z4, z8, 0);
}

// ---------------- prep kernels ----------------

__global__ void k_zero16(uint4* __restrict__ p, long n16) {
  long i = (long)blockIdx.x * blockDim.x + threadIdx.x;
  if (i < n16) { uint4 z; z.x = z.y = z.z = z.w = 0u; p[i] = z; }
}

__global__ void k_cvt_bf16(const float* __restrict__ in, __bf16* __restrict__ out, long n) {
  long i = (long)blockIdx.x * blockDim.x + threadIdx.x;
  if (i < n) out[i] = (__bf16)in[i];
}

// out[c*R + r] = (bf16) in[r*C + c]
__global__ void k_transpose_bf16(const float* __restrict__ in, __bf16* __restrict__ out,
                                 int R, int Cc) {
  int i = blockIdx.x * blockDim.x + threadIdx.x;
  if (i < R * Cc) {
    int r = i / Cc, c = i % Cc;
    out[(size_t)c * R + r] = (__bf16)in[i];
  }
}

// bias[h][n][m] = table[idx[n*196+m]*12 + h]
__global__ void k_bias(const float* __restrict__ table, const int* __restrict__ idx,
                       float* __restrict__ bias) {
  int i = blockIdx.x * blockDim.x + threadIdx.x;
  if (i < H_ * N_ * N_) {
    int h = i / (N_ * N_);
    int rem = i % (N_ * N_);
    bias[i] = table[idx[rem] * H_ + h];
  }
}

// ---------------- GEMM1: qkv = x @ W_qkv + b, scatter to q/k/vT ----------------
// grid (RT/16, 36), block 256 (8 waves); wave: 32 rows x 64 cols, K=768 in steps of 32.
__global__ void k_gemm_qkv(const __bf16* __restrict__ xb, const __bf16* __restrict__ wt,
                           const float* __restrict__ qb_bias,
                           __bf16* __restrict__ Q, __bf16* __restrict__ K,
                           __bf16* __restrict__ VT) {
  const int lane = threadIdx.x & 31;
  const int wave = threadIdx.x >> 5;
  const int la = lane & 15;
  const int hi = (lane >> 4) & 1;
  const int m0 = (blockIdx.x * 8 + wave) * 32;
  const int c0 = blockIdx.y * 64;

  f32x8 acc[2][4];
#pragma unroll
  for (int rr = 0; rr < 2; ++rr)
#pragma unroll
    for (int s = 0; s < 4; ++s)
#pragma unroll
      for (int r = 0; r < 8; ++r) acc[rr][s][r] = 0.f;

  const __bf16* arow0 = xb + (size_t)(m0 + la) * C_;
  const __bf16* arow1 = xb + (size_t)(m0 + 16 + la) * C_;
  for (int kb = 0; kb < C_; kb += 32) {
    FragBF a0, a1;
    const __bf16* ap0 = arow0 + kb + hi * 8;
    const __bf16* ap1 = arow1 + kb + hi * 8;
    a0.q[0] = *(const uint4*)(ap0); a0.q[1] = *(const uint4*)(ap0 + 16);
    a1.q[0] = *(const uint4*)(ap1); a1.q[1] = *(const uint4*)(ap1 + 16);
#pragma unroll
    for (int s = 0; s < 4; ++s) {
      FragBF bf;
      const __bf16* bp = wt + (size_t)(c0 + s * 16 + la) * C_ + kb + hi * 8;
      bf.q[0] = *(const uint4*)(bp);
      bf.q[1] = *(const uint4*)(bp + 16);
      acc[0][s] = wmma_bf16(a0, bf, acc[0][s]);
      acc[1][s] = wmma_bf16(a1, bf, acc[1][s]);
    }
  }

#pragma unroll
  for (int s = 0; s < 4; ++s) {
    const int gcol = c0 + s * 16 + la;
    const int which = gcol / C_;
    const int rem = gcol % C_;
    const int h = rem / DH;
    const int dd = rem % DH;
    const float badd = qb_bias[gcol];
#pragma unroll
    for (int rr = 0; rr < 2; ++rr) {
#pragma unroll
      for (int r = 0; r < 8; ++r) {
        const int grow = m0 + rr * 16 + r + hi * 8;
        const int bb = grow / N_;
        const int n = grow % N_;
        const float val = acc[rr][s][r] + badd;
        const size_t head = (size_t)(bb * H_ + h);
        if (which == 0)      Q[(head * NPAD + n) * DH + dd] = (__bf16)(val * 0.125f);
        else if (which == 1) K[(head * NPAD + n) * DH + dd] = (__bf16)val;
        else                 VT[(head * DH + dd) * KPAD + n] = (__bf16)val;
      }
    }
  }
}

// ---------------- Attention: one block per (b,h), K/V staged in LDS via TDM ----------------
__global__ void k_attention(const __bf16* __restrict__ Qg, const __bf16* __restrict__ Kg,
                            const __bf16* __restrict__ VTg, const float* __restrict__ BIAS,
                            __bf16* __restrict__ OUT) {
  extern __shared__ __bf16 smem[];
  __bf16* sP = smem;                       // [NPAD][KPAD] = 93,184 B
  __bf16* sK = smem + NPAD * KPAD;         // [NPAD][DH]   = 26,624 B
  __bf16* sV = sK + NPAD * DH;             // [DH][KPAD]   = 28,672 B
  const int bh = blockIdx.x;
  const int b = bh / H_, h = bh % H_;
  const int lane = threadIdx.x & 31, wave = threadIdx.x >> 5;
  const int la = lane & 15, hi = (lane >> 4) & 1;
  const __bf16* qb = Qg + (size_t)bh * NPAD * DH;
  const __bf16* kb = Kg + (size_t)bh * NPAD * DH;
  const __bf16* vb = VTg + (size_t)bh * DH * KPAD;
  const float* bbias = BIAS + (size_t)h * N_ * N_;

  // TDM: one wave kicks off K and V tile DMAs (TENSORcnt-tracked, EXEC-ignored)
  if (wave == 0) {
    tdm_load_1d_bf16(sK, kb, NPAD * DH);
    tdm_load_1d_bf16(sV, vb, DH * KPAD);
  }
  // meanwhile every thread zeroes sP (pad cols 208..223 must be 0)
  {
    uint4 z; z.x = z.y = z.z = z.w = 0u;
    for (int i = threadIdx.x; i < NPAD * KPAD / 8; i += blockDim.x)
      ((uint4*)sP)[i] = z;
  }
  __builtin_amdgcn_s_wait_tensorcnt(0);    // trivially passes for waves 1..7
  __syncthreads();

  // phase 1: S = (Q/8) K^T + bias, bf16 into LDS, masked cols/rows -> -1e30
  for (int t = wave; t < 13 * 13; t += 8) {
    const int tr = t / 13, tc = t % 13;
    f32x8 acc;
#pragma unroll
    for (int r = 0; r < 8; ++r) acc[r] = 0.f;
    const __bf16* ar = qb + (size_t)(tr * 16 + la) * DH;
    const __bf16* br = sK + (size_t)(tc * 16 + la) * DH;
#pragma unroll
    for (int kk = 0; kk < DH; kk += 32) {
      FragBF a, bf;
      const __bf16* ap = ar + kk + hi * 8;
      a.q[0] = *(const uint4*)ap;
      a.q[1] = *(const uint4*)(ap + 16);
      const __bf16* bp = br + kk + hi * 8;
      bf.q[0] = *(const uint4*)bp;
      bf.q[1] = *(const uint4*)(bp + 16);
      acc = wmma_bf16(a, bf, acc);
    }
#pragma unroll
    for (int r = 0; r < 8; ++r) {
      const int m = tr * 16 + r + hi * 8;
      const int n = tc * 16 + la;
      const float s = (m < N_ && n < N_) ? (acc[r] + bbias[m * N_ + n]) : -1e30f;
      sP[m * KPAD + n] = (__bf16)s;
    }
  }
  __syncthreads();

  // phase 2: row softmax in place (one wave per row; 7 elems/lane covers 224)
  for (int row = wave; row < NPAD; row += 8) {
    float v[7];
#pragma unroll
    for (int j = 0; j < 7; ++j) {
      const int c = j * 32 + lane;
      v[j] = (c < N_ && row < N_) ? (float)sP[row * KPAD + c] : -1e30f;
    }
    float mx = v[0];
#pragma unroll
    for (int j = 1; j < 7; ++j) mx = fmaxf(mx, v[j]);
    for (int o = 16; o; o >>= 1) mx = fmaxf(mx, __shfl_xor(mx, o));
    float sum = 0.f;
#pragma unroll
    for (int j = 0; j < 7; ++j) { v[j] = __expf(v[j] - mx); sum += v[j]; }
    for (int o = 16; o; o >>= 1) sum += __shfl_xor(sum, o);
    const float inv = 1.f / sum;
#pragma unroll
    for (int j = 0; j < 7; ++j) {
      const int c = j * 32 + lane;
      sP[row * KPAD + c] = (__bf16)((c < N_) ? v[j] * inv : 0.f);
    }
  }
  __syncthreads();

  // phase 3: O = P @ V  (K padded to 224; padded P is exactly 0)
  for (int t = wave; t < 13 * 4; t += 8) {
    const int tr = t / 4, tc = t % 4;
    f32x8 acc;
#pragma unroll
    for (int r = 0; r < 8; ++r) acc[r] = 0.f;
#pragma unroll
    for (int kk = 0; kk < KPAD; kk += 32) {
      FragBF a, bf;
      const __bf16* ap = sP + (size_t)(tr * 16 + la) * KPAD + kk + hi * 8;
      a.q[0] = *(const uint4*)ap;
      a.q[1] = *(const uint4*)(ap + 16);
      const __bf16* bp = sV + (size_t)(tc * 16 + la) * KPAD + kk + hi * 8;
      bf.q[0] = *(const uint4*)bp;
      bf.q[1] = *(const uint4*)(bp + 16);
      acc = wmma_bf16(a, bf, acc);
    }
#pragma unroll
    for (int r = 0; r < 8; ++r) {
      const int m = tr * 16 + r + hi * 8;
      if (m < N_) {
        const int dd = tc * 16 + la;
        OUT[((size_t)b * N_ + m) * C_ + h * DH + dd] = (__bf16)acc[r];
      }
    }
  }
}

// ---------------- GEMM2: out = attn_out @ W_proj + b (f32 out) ----------------
__global__ void k_gemm_proj(const __bf16* __restrict__ ain, const __bf16* __restrict__ wt,
                            const float* __restrict__ pb, float* __restrict__ out) {
  const int lane = threadIdx.x & 31;
  const int wave = threadIdx.x >> 5;
  const int la = lane & 15;
  const int hi = (lane >> 4) & 1;
  const int m0 = (blockIdx.x * 8 + wave) * 32;
  const int c0 = blockIdx.y * 64;

  f32x8 acc[2][4];
#pragma unroll
  for (int rr = 0; rr < 2; ++rr)
#pragma unroll
    for (int s = 0; s < 4; ++s)
#pragma unroll
      for (int r = 0; r < 8; ++r) acc[rr][s][r] = 0.f;

  const __bf16* arow0 = ain + (size_t)(m0 + la) * C_;
  const __bf16* arow1 = ain + (size_t)(m0 + 16 + la) * C_;
  for (int kb = 0; kb < C_; kb += 32) {
    FragBF a0, a1;
    const __bf16* ap0 = arow0 + kb + hi * 8;
    const __bf16* ap1 = arow1 + kb + hi * 8;
    a0.q[0] = *(const uint4*)(ap0); a0.q[1] = *(const uint4*)(ap0 + 16);
    a1.q[0] = *(const uint4*)(ap1); a1.q[1] = *(const uint4*)(ap1 + 16);
#pragma unroll
    for (int s = 0; s < 4; ++s) {
      FragBF bf;
      const __bf16* bp = wt + (size_t)(c0 + s * 16 + la) * C_ + kb + hi * 8;
      bf.q[0] = *(const uint4*)(bp);
      bf.q[1] = *(const uint4*)(bp + 16);
      acc[0][s] = wmma_bf16(a0, bf, acc[0][s]);
      acc[1][s] = wmma_bf16(a1, bf, acc[1][s]);
    }
  }

#pragma unroll
  for (int s = 0; s < 4; ++s) {
    const int col = c0 + s * 16 + la;
    const float badd = pb[col];
#pragma unroll
    for (int rr = 0; rr < 2; ++rr) {
#pragma unroll
      for (int r = 0; r < 8; ++r) {
        const int row = m0 + rr * 16 + r + hi * 8;
        out[(size_t)row * C_ + col] = acc[rr][s][r] + badd;
      }
    }
  }
}

// ---------------- launch ----------------
extern "C" void kernel_launch(void* const* d_in, const int* in_sizes, int n_in,
                              void* d_out, int out_size, void* d_ws, size_t ws_size,
                              hipStream_t stream) {
  (void)in_sizes; (void)n_in; (void)out_size; (void)ws_size;
  const float* x         = (const float*)d_in[0];
  const float* qkv_w     = (const float*)d_in[1];
  const float* qkv_b     = (const float*)d_in[2];
  const float* proj_w    = (const float*)d_in[3];
  const float* proj_b    = (const float*)d_in[4];
  const float* rpe_table = (const float*)d_in[5];
  const int*   rpe_index = (const int*)d_in[6];
  float* out = (float*)d_out;
  char* ws = (char*)d_ws;

  size_t off = 0;
  auto alloc = [&](size_t bytes) { size_t o = off; off = (off + bytes + 255) & ~(size_t)255; return o; };
  __bf16* xbf    = (__bf16*)(ws + alloc((size_t)ROWS * C_ * 2));
  __bf16* qkvwt  = (__bf16*)(ws + alloc((size_t)3 * C_ * C_ * 2));
  __bf16* projwt = (__bf16*)(ws + alloc((size_t)C_ * C_ * 2));
  __bf16* Q      = (__bf16*)(ws + alloc((size_t)BH * NPAD * DH * 2));
  __bf16* K      = (__bf16*)(ws + alloc((size_t)BH * NPAD * DH * 2));
  __bf16* VT     = (__bf16*)(ws + alloc((size_t)BH * DH * KPAD * 2));
  float*  bias   = (float*)(ws + alloc((size_t)H_ * N_ * N_ * 4));
  __bf16* aout   = xbf;  // reuse: x_bf fully consumed by GEMM1 before attention writes here

  {
    long n16 = ((size_t)BH * NPAD * DH * 2 * 2 + (size_t)BH * DH * KPAD * 2) / 16;
    k_zero16<<<(unsigned)((n16 + 255) / 256), 256, 0, stream>>>((uint4*)Q, n16);
  }
  {
    long n = (long)ROWS * C_;
    k_cvt_bf16<<<(unsigned)((n + 255) / 256), 256, 0, stream>>>(x, xbf, n);
  }
  k_transpose_bf16<<<(C_ * 3 * C_ + 255) / 256, 256, 0, stream>>>(qkv_w, qkvwt, C_, 3 * C_);
  k_transpose_bf16<<<(C_ * C_ + 255) / 256, 256, 0, stream>>>(proj_w, projwt, C_, C_);
  k_bias<<<(H_ * N_ * N_ + 255) / 256, 256, 0, stream>>>(rpe_table, rpe_index, bias);

  k_gemm_qkv<<<dim3(RT / 16, 36), 256, 0, stream>>>(xbf, qkvwt, qkv_b, Q, K, VT);

  const unsigned shmem = (NPAD * KPAD + NPAD * DH + DH * KPAD) * 2;  // 148,480 B
  k_attention<<<BH, 256, shmem, stream>>>(Q, K, VT, bias, aout);

  k_gemm_proj<<<dim3(RT / 16, 12), 256, 0, stream>>>(aout, projwt, proj_b, out);
}
